// EdgeAngleConv_82463372083468
// MI455X (gfx1250) — matrix operand order repaired and checked by reading, not compiled, not run on hardware
//
#include <hip/hip_runtime.h>
#include <hip/hip_bf16.h>

// ---------------------------------------------------------------------------
// EdgeAngleConv for MI455X (gfx1250, wave32, WMMA).
//
// Roofline: ~32 GFLOP of GEMM (node MLP hoisted out of the edge loop),
// ~0.75 GB fundamental traffic (h_n gather 307MB + f32 atomic scatter 307MB
// dominate) -> ~32us floor @ 23.3 TB/s. f16-in/f32-acc WMMA keeps GEMM time
// under that floor. This round: 64 edges/block with register-resident B
// fragments (4x less weight re-fetch), and global_prefetch of the randomly
// gathered h_n rows so HBM latency hides behind the two GEMM stages.
// ---------------------------------------------------------------------------

typedef __attribute__((ext_vector_type(16))) _Float16 v16h;
typedef __attribute__((ext_vector_type(2)))  _Float16 v2h;
typedef __attribute__((ext_vector_type(8)))  float    v8f;

#define HID 128
#define EDGE_DIM 32
#define REL_DIM 16
#define EFPAD 64          // 48 padded to 64 -> two K=32 WMMA steps
#define LN_EPS 1e-5f

// ---- WMMA fragment loaders (layouts per CDNA5 ISA 7.12.2) ------

// A fragment (16x32, f16, row-major source, leading dim `ld`):
// lane l: M = l%16, g = l/16; VGPR v pair p holds K = (v>=4?16:0)+8g+(v&3)*2+p
__device__ __forceinline__ v16h load_a_frag(const _Float16* base, int ld, int k0, int lane) {
    const int m = lane & 15, g = lane >> 4;
    const _Float16* row = base + (size_t)m * ld + k0 + 8 * g;
    v16h a;
#pragma unroll
    for (int v = 0; v < 8; ++v) {
        const int kb = ((v >= 4) ? 16 : 0) + ((v & 3) * 2);
        v2h pr = *(const v2h*)(row + kb);
        a[2 * v]     = pr[0];
        a[2 * v + 1] = pr[1];
    }
    return a;
}

// B fragment (32x16, f16). Weights stored TRANSPOSED as wt[N][K] so each
// per-lane (K=2v,2v+1) pair is one contiguous dword load.
// lane l: N = l%16, g = l/16; VGPR v pair p holds K = (v>=4?16:0)+8g+(v&3)*2+p.
__device__ __forceinline__ v16h load_b_frag(const _Float16* wt, int ldk, int c0, int k0, int lane) {
    const int n = lane & 15, g = lane >> 4;
    const _Float16* col = wt + (size_t)(c0 + n) * ldk + k0 + 8 * g;
    v16h b;
#pragma unroll
    for (int v = 0; v < 8; ++v) {
        v2h pr = *(const v2h*)(col + ((v >= 4) ? 16 : 0) + ((v & 3) * 2));
        b[2 * v]     = pr[0];
        b[2 * v + 1] = pr[1];
    }
    return b;
}

__device__ __forceinline__ v8f wmma16(v16h a, v16h b, v8f c) {
    return __builtin_amdgcn_wmma_f32_16x16x32_f16(false, a, false, b, (short)0, c, false, false);
}

// Native fp32 global atomic add (GLOBAL_ATOMIC_ADD_F32, no CAS loop).
__device__ __forceinline__ void atomic_add_f32(float* p, float v) {
    asm volatile("global_atomic_add_f32 %0, %1, off" : : "v"(p), "v"(v) : "memory");
}

// ---------------------------- prep kernels ---------------------------------

__global__ void k_f32_to_f16(const float* __restrict__ s, _Float16* __restrict__ d, int n) {
    int i = blockIdx.x * blockDim.x + threadIdx.x;
    if (i < n) d[i] = (_Float16)s[i];
}

// wt[n*Kpad + k] = (k < K) ? w[k*HID + n] : 0   (w is [K, HID] row-major)
__global__ void k_transpose_f16(const float* __restrict__ w, _Float16* __restrict__ wt,
                                int K, int Kpad) {
    int idx = blockIdx.x * blockDim.x + threadIdx.x;
    int total = HID * Kpad;
    if (idx >= total) return;
    int n = idx / Kpad, k = idx % Kpad;
    wt[idx] = (k < K) ? (_Float16)w[(size_t)k * HID + n] : (_Float16)0.f;
}

// ----------------------- node MLP: hn = relu(x@Wn+bn) ----------------------
// 8 waves/block, wave = column tile. Block covers 4 row strips with B frags
// held in registers across strips (weight loads amortized 4x).
__global__ void k_node_mlp(const _Float16* __restrict__ xh, const _Float16* __restrict__ wtn,
                           const float* __restrict__ bn, _Float16* __restrict__ hn, int ntiles) {
    const int tile0 = blockIdx.x * 4;
    const int wave = threadIdx.x >> 5, lane = threadIdx.x & 31;
    const int c0 = wave * 16, g = lane >> 4, col = c0 + (lane & 15);

    v16h bw[4];
#pragma unroll
    for (int kc = 0; kc < 4; ++kc) bw[kc] = load_b_frag(wtn, HID, c0, kc * 32, lane);
    const float bias = bn[col];

#pragma unroll
    for (int rt = 0; rt < 4; ++rt) {
        const int tile = tile0 + rt;
        if (tile >= ntiles) break;
        const int row0 = tile * 16;
        const _Float16* A = xh + (size_t)row0 * HID;
        v8f acc = {};
#pragma unroll
        for (int kc = 0; kc < 4; ++kc)
            acc = wmma16(load_a_frag(A, HID, kc * 32, lane), bw[kc], acc);
#pragma unroll
        for (int r = 0; r < 8; ++r) {
            float v = acc[r] + bias;
            v = v > 0.f ? v : 0.f;
            hn[(size_t)(row0 + r + 8 * g) * HID + col] = (_Float16)v;
        }
    }
}

// -------------------------- edge MLP + scatter -----------------------------
// Block = 64 edges (4 sub-tiles of 16). B fragments for both MLP stages are
// register-resident for the whole block. h_n rows are prefetched as soon as
// src indices are known so the random gather hides behind the GEMMs.
__global__ void k_edge(const float* __restrict__ edge_attr, const int* __restrict__ edge_index,
                       const int* __restrict__ edge_type, const float* __restrict__ rel_embed,
                       const float* __restrict__ conf,
                       const _Float16* __restrict__ wte1, const float* __restrict__ be1,
                       const _Float16* __restrict__ wte2, const float* __restrict__ be2,
                       const _Float16* __restrict__ hn, float* __restrict__ aggr, int E) {
    __shared__ _Float16 s_ef[64 * EFPAD];     // 8 KB
    __shared__ _Float16 s_m1[64 * HID];       // 16 KB
    __shared__ int   s_src[64];
    __shared__ int   s_dst[64];
    __shared__ float s_conf[64];

    const int t = threadIdx.x;
    const int e0 = blockIdx.x * 64;

    if (t < 64) {
        const int e = e0 + t;
        const int s = edge_index[e];
        s_src[t] = s;
        s_dst[t] = edge_index[E + e];
        s_conf[t] = conf[s];
        // prefetch both 128B cachelines of the h_n row we will gather later
        const _Float16* hrow = hn + (size_t)s * HID;
        __builtin_prefetch(hrow, 0, 1);
        __builtin_prefetch(hrow + 64, 0, 1);
    }
    // build ef = [edge_attr | rel_embed[edge_type] | 0-pad] as f16 in LDS
    for (int idx = t; idx < 64 * EFPAD; idx += 256) {
        const int el = idx >> 6, c = idx & (EFPAD - 1);
        float v = 0.f;
        if (c < EDGE_DIM)
            v = edge_attr[(size_t)(e0 + el) * EDGE_DIM + c];
        else if (c < EDGE_DIM + REL_DIM)
            v = rel_embed[(size_t)edge_type[e0 + el] * REL_DIM + (c - EDGE_DIM)];
        s_ef[idx] = (_Float16)v;
    }

    const int wave = t >> 5, lane = t & 31;
    const int c0 = wave * 16, g = lane >> 4, col = c0 + (lane & 15);

    // register-resident B fragments for both stages
    v16h b1[2], b2[4];
#pragma unroll
    for (int kc = 0; kc < 2; ++kc) b1[kc] = load_b_frag(wte1, EFPAD, c0, kc * 32, lane);
#pragma unroll
    for (int kc = 0; kc < 4; ++kc) b2[kc] = load_b_frag(wte2, HID, c0, kc * 32, lane);
    const float bias1 = be1[col];
    const float bias2 = be2[col];

    __syncthreads();

    // stage 1: msg1 = relu(ef @ We1 + b1) for all 4 sub-tiles
#pragma unroll
    for (int st = 0; st < 4; ++st) {
        const _Float16* A = s_ef + st * 16 * EFPAD;
        v8f acc = {};
        acc = wmma16(load_a_frag(A, EFPAD, 0,  lane), b1[0], acc);
        acc = wmma16(load_a_frag(A, EFPAD, 32, lane), b1[1], acc);
#pragma unroll
        for (int r = 0; r < 8; ++r) {
            float v = acc[r] + bias1;
            v = v > 0.f ? v : 0.f;
            s_m1[(st * 16 + r + 8 * g) * HID + col] = (_Float16)v;
        }
    }
    __syncthreads();

    // stage 2: msg = msg1 @ We2 + b2 + hn[src]; scale by conf[src]; scatter
#pragma unroll
    for (int st = 0; st < 4; ++st) {
        const _Float16* A = s_m1 + st * 16 * HID;
        v8f acc = {};
#pragma unroll
        for (int kc = 0; kc < 4; ++kc)
            acc = wmma16(load_a_frag(A, HID, kc * 32, lane), b2[kc], acc);
#pragma unroll
        for (int r = 0; r < 8; ++r) {
            const int m = st * 16 + r + 8 * g;          // edge row within block
            const int src = s_src[m], dst = s_dst[m];
            float v = acc[r] + bias2 + (float)hn[(size_t)src * HID + col];
            v *= s_conf[m];
            atomic_add_f32(&aggr[(size_t)dst * HID + col], v);
        }
    }
}

// ------------------- gate = sigmoid([x|aggr] @ Wg + bg) --------------------
__global__ void k_gate(const _Float16* __restrict__ xh, const _Float16* __restrict__ aggrh,
                       const _Float16* __restrict__ wtg, const float* __restrict__ bg,
                       float* __restrict__ gate, int ntiles) {
    const int tile0 = blockIdx.x * 4;
    const int wave = threadIdx.x >> 5, lane = threadIdx.x & 31;
    const int c0 = wave * 16, g = lane >> 4, col = c0 + (lane & 15);

    v16h bw[8];
#pragma unroll
    for (int kc = 0; kc < 8; ++kc) bw[kc] = load_b_frag(wtg, 2 * HID, c0, kc * 32, lane);
    const float b = bg[col];

#pragma unroll
    for (int rt = 0; rt < 4; ++rt) {
        const int tile = tile0 + rt;
        if (tile >= ntiles) break;
        const int row0 = tile * 16;
        v8f acc = {};
#pragma unroll
        for (int kc = 0; kc < 8; ++kc) {
            const _Float16* A = (kc < 4) ? (xh + (size_t)row0 * HID)
                                         : (aggrh + (size_t)row0 * HID);
            const int ka = (kc < 4) ? kc * 32 : (kc - 4) * 32;
            acc = wmma16(load_a_frag(A, HID, ka, lane), bw[kc], acc);
        }
#pragma unroll
        for (int r = 0; r < 8; ++r) {
            const float z = acc[r] + b;
            gate[(size_t)(row0 + r + 8 * g) * HID + col] = 1.f / (1.f + expf(-z));
        }
    }
}

// ---------------- fused = g*tanh(aggr)+(1-g)*x; LayerNorm ------------------
__global__ void k_fuse_ln(const float* __restrict__ x, const float* __restrict__ aggr,
                          const float* __restrict__ gate, const float* __restrict__ gamma,
                          const float* __restrict__ beta, float* __restrict__ out) {
    const int n = blockIdx.x, c = threadIdx.x;
    const size_t o = (size_t)n * HID + c;
    const float a = aggr[o], xx = x[o], gt = gate[o];
    const float fused = gt * tanhf(a) + (1.f - gt) * xx;

    __shared__ float red[HID];
    red[c] = fused;
    __syncthreads();
#pragma unroll
    for (int s = HID / 2; s > 0; s >>= 1) {
        if (c < s) red[c] += red[c + s];
        __syncthreads();
    }
    const float mu = red[0] * (1.f / HID);
    __syncthreads();
    const float d = fused - mu;
    red[c] = d * d;
    __syncthreads();
#pragma unroll
    for (int s = HID / 2; s > 0; s >>= 1) {
        if (c < s) red[c] += red[c + s];
        __syncthreads();
    }
    const float var = red[0] * (1.f / HID);
    out[o] = d * rsqrtf(var + LN_EPS) * gamma[c] + beta[c];
}

// ------------------------------- launcher ----------------------------------

extern "C" void kernel_launch(void* const* d_in, const int* in_sizes, int n_in,
                              void* d_out, int out_size, void* d_ws, size_t ws_size,
                              hipStream_t stream) {
    const int N = in_sizes[0] / HID;        // 50000
    const int E = in_sizes[3];              // 600000 (edge_type)

    const float* x          = (const float*)d_in[0];
    const int*   edge_index = (const int*)  d_in[1];
    const float* edge_attr  = (const float*)d_in[2];
    const int*   edge_type  = (const int*)  d_in[3];
    const float* confidence = (const float*)d_in[4];
    const float* rel_embed  = (const float*)d_in[5];
    const float* W_e1 = (const float*)d_in[6];
    const float* b_e1 = (const float*)d_in[7];
    const float* W_e2 = (const float*)d_in[8];
    const float* b_e2 = (const float*)d_in[9];
    const float* W_n  = (const float*)d_in[10];
    const float* b_n  = (const float*)d_in[11];
    const float* W_g  = (const float*)d_in[12];
    const float* b_g  = (const float*)d_in[13];
    const float* ln_g = (const float*)d_in[14];
    const float* ln_b = (const float*)d_in[15];
    float* out = (float*)d_out;

    // workspace carve-up (all offsets multiple of 256B)
    char* ws = (char*)d_ws;
    size_t off = 0;
    auto take = [&](size_t bytes) { char* p = ws + off; off += (bytes + 255) & ~(size_t)255; return p; };
    _Float16* xh    = (_Float16*)take((size_t)N * HID * 2);
    _Float16* hn    = (_Float16*)take((size_t)N * HID * 2);
    float*    aggr  = (float*)   take((size_t)N * HID * 4);
    _Float16* aggrh = (_Float16*)take((size_t)N * HID * 2);
    float*    gate  = (float*)   take((size_t)N * HID * 4);
    _Float16* wtn   = (_Float16*)take((size_t)HID * HID * 2);
    _Float16* wte1  = (_Float16*)take((size_t)HID * EFPAD * 2);
    _Float16* wte2  = (_Float16*)take((size_t)HID * HID * 2);
    _Float16* wtg   = (_Float16*)take((size_t)HID * 2 * HID * 2);
    (void)ws_size;

    hipMemsetAsync(aggr, 0, (size_t)N * HID * 4, stream);

    const int nxh = N * HID;
    k_f32_to_f16<<<(nxh + 255) / 256, 256, 0, stream>>>(x, xh, nxh);
    k_transpose_f16<<<(HID * HID   + 255) / 256, 256, 0, stream>>>(W_n,  wtn,  HID, HID);
    k_transpose_f16<<<(HID * EFPAD + 255) / 256, 256, 0, stream>>>(W_e1, wte1, EDGE_DIM + REL_DIM, EFPAD);
    k_transpose_f16<<<(HID * HID   + 255) / 256, 256, 0, stream>>>(W_e2, wte2, HID, HID);
    k_transpose_f16<<<(HID * 2*HID + 255) / 256, 256, 0, stream>>>(W_g,  wtg,  2 * HID, 2 * HID);

    const int ntiles = N / 16;                       // 3125 row tiles
    const int nblocks = (ntiles + 3) / 4;            // 4 tiles per block
    k_node_mlp<<<nblocks, 256, 0, stream>>>(xh, wtn, b_n, hn, ntiles);

    k_edge<<<E / 64, 256, 0, stream>>>(edge_attr, edge_index, edge_type, rel_embed, confidence,
                                       wte1, b_e1, wte2, b_e2, hn, aggr, E);

    k_f32_to_f16<<<(nxh + 255) / 256, 256, 0, stream>>>(aggr, aggrh, nxh);

    k_gate<<<nblocks, 256, 0, stream>>>(xh, aggrh, wtg, b_g, gate, ntiles);

    k_fuse_ln<<<N, HID, 0, stream>>>(x, aggr, gate, ln_g, ln_b, out);
    (void)out_size; (void)n_in;
}
